// HyperTopoGML_Backbone_29695403884555
// MI455X (gfx1250) — compile-verified, matrix operand order checked by reading, not run on hardware
//
#include <hip/hip_runtime.h>
#include <cstdint>
#include <cstddef>

// ---------------- types ----------------
typedef __attribute__((ext_vector_type(16))) __bf16 bfvec16;
typedef __attribute__((ext_vector_type(8)))  float  fvec8;
typedef unsigned short u16;
typedef unsigned int   u32;

union Frag { bfvec16 v; u32 u[8]; };

#define NN   100000
#define NE   500000
#define DD   128
#define VV   3
#define HIDN 145
#define ABW  304          // padded 2*HIDN (19 tiles of 16)
#define EPS_ 1e-15f

// ---------------- device helpers ----------------
__device__ __forceinline__ float wsum(float v) {
#pragma unroll
  for (int i = 16; i > 0; i >>= 1) v += __shfl_xor(v, i, 32);
  return v;
}
__device__ __forceinline__ float artanh_(float x) {
  return 0.5f * logf((1.f + x) / (1.f - x));
}
__device__ __forceinline__ u16 f2bf(float x) {           // RNE f32 -> bf16
  u32 u = __float_as_uint(x);
  u32 r = u + 0x7FFFu + ((u >> 16) & 1u);
  return (u16)(r >> 16);
}
__device__ __forceinline__ float bf2f(u16 h) { return __uint_as_float(((u32)h) << 16); }

// ---------------- weight prep kernels ----------------
// Build W1ab bf16 hi/lo: rows 0..144 = W1[:, :128], rows 145..289 = W1[:, 128:256]
__global__ __launch_bounds__(256) void conv_w1ab(const float* __restrict__ W1v,
                                                 u16* __restrict__ wh, u16* __restrict__ wl) {
  int i = blockIdx.x * 256 + threadIdx.x;
  if (i >= ABW * DD) return;
  int r = i >> 7, k = i & 127;
  float v = 0.f;
  if (r < HIDN)            v = W1v[r * 291 + k];
  else if (r < 2 * HIDN)   v = W1v[(r - HIDN) * 291 + 128 + k];
  u16 h = f2bf(v);
  wh[i] = h;
  wl[i] = f2bf(v - bf2f(h));
}

__global__ __launch_bounds__(256) void conv_w128(const float* __restrict__ W,
                                                 u16* __restrict__ wh, u16* __restrict__ wl) {
  int i = blockIdx.x * 256 + threadIdx.x;
  if (i >= DD * DD) return;
  float v = W[i];
  u16 h = f2bf(v);
  wh[i] = h;
  wl[i] = f2bf(v - bf2f(h));
}

// consts layout (floats): [0..303] cvec, [304..463] bias0, [464..623] w2c, [624] b2
__global__ __launch_bounds__(320) void rewire_consts(const float* __restrict__ W1v,
                                                     const float* __restrict__ b1v,
                                                     const float* __restrict__ W2v,
                                                     const float* __restrict__ b2v,
                                                     const float* __restrict__ ctab,
                                                     const int* __restrict__ ctid,
                                                     float* __restrict__ consts) {
  int t = threadIdx.x;
  int cid = *ctid;
  const float* cemb = ctab + cid * 32;
  if (t < ABW) {
    float cv = 0.f;
    if (t < HIDN)           cv = W1v[t * 291 + 288] + W1v[t * 291 + 290];
    else if (t < 2 * HIDN)  cv = W1v[(t - HIDN) * 291 + 289] - W1v[(t - HIDN) * 291 + 290];
    consts[t] = cv;
  }
  if (t < HIDN) {
    float b = b1v[t];
#pragma unroll 4
    for (int j = 0; j < 32; ++j) b += W1v[t * 291 + 256 + j] * cemb[j];
    consts[304 + t] = b;
    consts[464 + t] = W2v[t];
  }
  if (t == 0) consts[624] = b2v[0];
}

// ---------------- bf16x3 WMMA GEMM:  Out[M,ldo] = X[M,128] @ W[ldo,128]^T ----------------
// Block = 256 threads = 8 waves; block computes 128 rows x 16 cols; K=128 in 4 chunks of 32.
__global__ __launch_bounds__(256) void gemm_bf16x3(
    const u16* __restrict__ Xh, const u16* __restrict__ Xl,
    const u16* __restrict__ Wh, const u16* __restrict__ Wl,
    float* __restrict__ Out, int M, int ldo,
    const float* __restrict__ cvec, const float* __restrict__ cscore) {
  __shared__ u16 Bh[16 * 130];
  __shared__ u16 Bl[16 * 130];
  __shared__ u16 Ah[128 * 36];
  __shared__ u16 Al[128 * 36];

  const int t = threadIdx.x;
  const int nBase = blockIdx.y * 16;
  const int rowBlock = blockIdx.x * 128;

  // Stage full weight tile (16 cols x 128 k), hi+lo, into LDS (coalesced 16B loads).
  {
    int n = t >> 4, seg = t & 15;
    const uint4* ph = (const uint4*)(Wh + (((size_t)(nBase + n)) << 7) + seg * 8);
    const uint4* pl = (const uint4*)(Wl + (((size_t)(nBase + n)) << 7) + seg * 8);
    uint4 vh = *ph, vl = *pl;
    u32* dh = (u32*)&Bh[n * 130 + seg * 8];
    u32* dl = (u32*)&Bl[n * 130 + seg * 8];
    dh[0] = vh.x; dh[1] = vh.y; dh[2] = vh.z; dh[3] = vh.w;
    dl[0] = vl.x; dl[1] = vl.y; dl[2] = vl.z; dl[3] = vl.w;
  }

  const int wave = t >> 5, lane = t & 31;
  const int lrow = lane & 15, lhi = lane >> 4;
  fvec8 acc = {0.f, 0.f, 0.f, 0.f, 0.f, 0.f, 0.f, 0.f};

#pragma unroll
  for (int kc = 0; kc < 4; ++kc) {
    __syncthreads();
    // Stage A chunk: 128 rows x 32 k, hi+lo (2 threads per row, 16 elems each).
    {
      int row = t >> 1, half = t & 1;
      int g = rowBlock + row;
      if (g >= M) g = M - 1;
      const uint4* ph = (const uint4*)(Xh + (((size_t)g) << 7) + kc * 32 + half * 16);
      const uint4* pl = (const uint4*)(Xl + (((size_t)g) << 7) + kc * 32 + half * 16);
      uint4 vh0 = ph[0], vh1 = ph[1];
      uint4 vl0 = pl[0], vl1 = pl[1];
      u32* dh = (u32*)&Ah[row * 36 + half * 16];
      u32* dl = (u32*)&Al[row * 36 + half * 16];
      dh[0] = vh0.x; dh[1] = vh0.y; dh[2] = vh0.z; dh[3] = vh0.w;
      dh[4] = vh1.x; dh[5] = vh1.y; dh[6] = vh1.z; dh[7] = vh1.w;
      dl[0] = vl0.x; dl[1] = vl0.y; dl[2] = vl0.z; dl[3] = vl0.w;
      dl[4] = vl1.x; dl[5] = vl1.y; dl[6] = vl1.z; dl[7] = vl1.w;
    }
    __syncthreads();

    // Assemble fragments per CDNA5 16-bit A/B layouts (K-pairs per dword).
    Frag fah, fal, fbh, fbl;
    int mloc = wave * 16 + lrow;
#pragma unroll
    for (int j = 0; j < 8; ++j) {
      int kb = ((j & 4) << 2) + ((j & 3) << 1) + (lhi << 3);  // (j>=4)*16 + (j%3bits)*2 + half*8
      fah.u[j] = *(const u32*)&Ah[mloc * 36 + kb];
      fal.u[j] = *(const u32*)&Al[mloc * 36 + kb];
      fbh.u[j] = *(const u32*)&Bh[lrow * 130 + kc * 32 + kb];
      fbl.u[j] = *(const u32*)&Bl[lrow * 130 + kc * 32 + kb];
    }
    // bf16x3: xh*wh + xl*wh + xh*wl, fp32 accumulate
    acc = __builtin_amdgcn_wmma_f32_16x16x32_bf16(false, fah.v, false, fbh.v, (short)0, acc, false, false);
    acc = __builtin_amdgcn_wmma_f32_16x16x32_bf16(false, fal.v, false, fbh.v, (short)0, acc, false, false);
    acc = __builtin_amdgcn_wmma_f32_16x16x32_bf16(false, fah.v, false, fbl.v, (short)0, acc, false, false);
  }

  // Epilogue: C layout — acc[i] -> row = base + i + 8*(lane>=16), col = nBase + lane%16
  int ncol = nBase + lrow;
  int rbase = rowBlock + wave * 16 + (lhi << 3);
  float cv = cvec ? cvec[ncol] : 0.f;
#pragma unroll
  for (int i = 0; i < 8; ++i) {
    int row = rbase + i;
    if (row < M) {
      float v = acc[i];
      if (cvec) v += cscore[row] * cv;
      Out[(size_t)row * ldo + ncol] = v;
    }
  }
}

// ---------------- node pointwise kernels (1 wave / row) ----------------
// expmap0(x_proj) -> hyp0 ; x_proj -> bf16 hi/lo pair
__global__ __launch_bounds__(256) void init_nodes(const float* __restrict__ xp,
                                                  u16* __restrict__ xph, u16* __restrict__ xpl,
                                                  float* __restrict__ hyp0, int n) {
  int row = blockIdx.x * 8 + (threadIdx.x >> 5);
  if (row >= n) return;
  int lane = threadIdx.x & 31;
  size_t rb = (size_t)row * DD;
  float x[4]; float n2 = 0.f;
#pragma unroll
  for (int j = 0; j < 4; ++j) { x[j] = xp[rb + lane + 32 * j]; n2 += x[j] * x[j]; }
  n2 = wsum(n2);
  float r = sqrtf(fmaxf(n2, EPS_));
  float sc = tanhf(r) / r;
#pragma unroll
  for (int j = 0; j < 4; ++j) {
    size_t idx = rb + lane + 32 * j;
    hyp0[idx] = sc * x[j];
    u16 bh = f2bf(x[j]);
    xph[idx] = bh;
    xpl[idx] = f2bf(x[j] - bf2f(bh));
  }
}

// t = logmap0(H) -> bf16 hi/lo pair (GEMM input)
__global__ __launch_bounds__(256) void logmap_bf16(const float* __restrict__ H,
                                                   u16* __restrict__ th, u16* __restrict__ tl, int n) {
  int row = blockIdx.x * 8 + (threadIdx.x >> 5);
  if (row >= n) return;
  int lane = threadIdx.x & 31;
  size_t rb = (size_t)row * DD;
  float h[4]; float n2 = 0.f;
#pragma unroll
  for (int j = 0; j < 4; ++j) { h[j] = H[rb + lane + 32 * j]; n2 += h[j] * h[j]; }
  n2 = wsum(n2);
  float hn = sqrtf(fmaxf(n2, EPS_));
  float nc = fminf(fmaxf(hn, EPS_), 1.f - 1e-5f);
  float lm = artanh_(nc) / hn;
#pragma unroll
  for (int j = 0; j < 4; ++j) {
    size_t idx = rb + lane + 32 * j;
    float tv = lm * h[j];
    u16 bh = f2bf(tv);
    th[idx] = bh;
    tl[idx] = f2bf(tv - bf2f(bh));
  }
}

// Y (=t@W^T) -> msg = logmap0(expmap0(Y)) in place; also seed agg with self-loop msg
__global__ __launch_bounds__(256) void msg_transform(float* __restrict__ Y,
                                                     float* __restrict__ agg, int n) {
  int row = blockIdx.x * 8 + (threadIdx.x >> 5);
  if (row >= n) return;
  int lane = threadIdx.x & 31;
  size_t rb = (size_t)row * DD;
  float y[4]; float n2 = 0.f;
#pragma unroll
  for (int j = 0; j < 4; ++j) { y[j] = Y[rb + lane + 32 * j]; n2 += y[j] * y[j]; }
  n2 = wsum(n2);
  float r = sqrtf(fmaxf(n2, EPS_));
  float sc = tanhf(r) / r;
  float xh[4]; float m2 = 0.f;
#pragma unroll
  for (int j = 0; j < 4; ++j) { xh[j] = sc * y[j]; m2 += xh[j] * xh[j]; }
  m2 = wsum(m2);
  float hn = sqrtf(fmaxf(m2, EPS_));
  float nc = fminf(fmaxf(hn, EPS_), 1.f - 1e-5f);
  float lm = artanh_(nc) / hn;
#pragma unroll
  for (int j = 0; j < 4; ++j) {
    size_t idx = rb + lane + 32 * j;
    float m = lm * xh[j];
    Y[idx] = m;
    agg[idx] = m;     // self-loop term initializes agg (no memset needed)
  }
}

// h_next = mobius_add(expmap0(leakyrelu(agg)), identity); optional tan output
__global__ __launch_bounds__(256) void update_nodes(const float* __restrict__ agg,
                                                    const float* __restrict__ ident,
                                                    float* __restrict__ outH,
                                                    float* __restrict__ tanOut,
                                                    int ldt, int colOff, int n) {
  int row = blockIdx.x * 8 + (threadIdx.x >> 5);
  if (row >= n) return;
  int lane = threadIdx.x & 31;
  size_t rb = (size_t)row * DD;
  float l[4], y[4]; float n2 = 0.f;
#pragma unroll
  for (int j = 0; j < 4; ++j) {
    float a = agg[rb + lane + 32 * j];
    float t = (a > 0.f) ? a : 0.1f * a;
    l[j] = t; n2 += t * t;
    y[j] = ident[rb + lane + 32 * j];
  }
  n2 = wsum(n2);
  float r = sqrtf(fmaxf(n2, EPS_));
  float sc = tanhf(r) / r;
  float o[4]; float x2p = 0.f, y2p = 0.f, xyp = 0.f;
#pragma unroll
  for (int j = 0; j < 4; ++j) {
    o[j] = sc * l[j];
    x2p += o[j] * o[j];
    y2p += y[j] * y[j];
    xyp += o[j] * y[j];
  }
  float x2 = wsum(x2p), y2 = wsum(y2p), xy = wsum(xyp);
  float ca = 1.f + 2.f * xy + y2;
  float cb = 1.f - x2;
  float den = fmaxf(1.f + 2.f * xy + x2 * y2, EPS_);
  float h[4]; float m2 = 0.f;
#pragma unroll
  for (int j = 0; j < 4; ++j) {
    h[j] = (ca * o[j] + cb * y[j]) / den;
    m2 += h[j] * h[j];
    outH[rb + lane + 32 * j] = h[j];
  }
  if (tanOut) {
    m2 = wsum(m2);
    float hn = sqrtf(fmaxf(m2, EPS_));
    float nc = fminf(fmaxf(hn, EPS_), 1.f - 1e-5f);
    float lm = artanh_(nc) / hn;
    size_t tb = (size_t)row * ldt + colOff;
#pragma unroll
    for (int j = 0; j < 4; ++j) tanOut[tb + lane + 32 * j] = lm * h[j];
  }
}

// ---------------- edge kernels (1 wave / edge) ----------------
__global__ __launch_bounds__(256) void edge_rewire(const float* __restrict__ AB,
                                                   const float* __restrict__ consts,
                                                   const int* __restrict__ srcI,
                                                   const int* __restrict__ dstI,
                                                   const float* __restrict__ escore,
                                                   float* __restrict__ ew, int e_cnt) {
  int e = blockIdx.x * 8 + (threadIdx.x >> 5);
  if (e >= e_cnt) return;
  int lane = threadIdx.x & 31;
  int s = srcI[e], d = dstI[e];
  const float* arow = AB + (size_t)s * ABW;
  const float* brow = AB + (size_t)d * ABW + HIDN;
  float acc = 0.f;
#pragma unroll
  for (int j = 0; j < 5; ++j) {
    int k = lane + 32 * j;
    if (k < HIDN) {
      float x = arow[k] + brow[k] + consts[304 + k];
      x = fmaxf(x, 0.f);
      acc += x * consts[464 + k];
    }
  }
  acc = wsum(acc);
  if (lane == 0) {
    float z = acc + consts[624];
    float dyn = 1.f / (1.f + expf(-z));
    ew[e] = escore[e] * dyn;
  }
}

__global__ __launch_bounds__(256) void edge_scatter(const float* __restrict__ msg,
                                                    const int* __restrict__ srcI,
                                                    const int* __restrict__ dstI,
                                                    const float* __restrict__ ew,
                                                    float* __restrict__ agg, int e_cnt) {
  int e = blockIdx.x * 8 + (threadIdx.x >> 5);
  if (e >= e_cnt) return;
  int lane = threadIdx.x & 31;
  int s = srcI[e], d = dstI[e];
  float w = ew[e];
  const float* ms = msg + (size_t)s * DD;
  float* ag = agg + (size_t)d * DD;
#pragma unroll
  for (int j = 0; j < 4; ++j) {
    int c = lane + 32 * j;
    atomicAdd(&ag[c], ms[c] * w);
  }
}

// ---------------- host orchestration ----------------
extern "C" void kernel_launch(void* const* d_in, const int* in_sizes, int n_in,
                              void* d_out, int out_size, void* d_ws, size_t ws_size,
                              hipStream_t stream) {
  (void)in_sizes; (void)n_in; (void)out_size;
  const float* x_proj      = (const float*)d_in[0];
  const float* edge_scores = (const float*)d_in[1];
  const float* causal      = (const float*)d_in[2];
  const float* ctab        = (const float*)d_in[3];
  const float* rw_W1       = (const float*)d_in[4];
  const float* rw_b1       = (const float*)d_in[5];
  const float* rw_W2       = (const float*)d_in[6];
  const float* rw_b2       = (const float*)d_in[7];
  const float* g1W         = (const float*)d_in[8];
  const float* g2W         = (const float*)d_in[9];
  const int*   eidx        = (const int*)d_in[10];
  const int*   ctid        = (const int*)d_in[11];
  float* out = (float*)d_out;

  char* base = (char*)d_ws;
  size_t off = 0;
  auto take = [&](size_t bytes) -> void* {
    void* p = base + off;
    off += (bytes + 255) & ~(size_t)255;
    return p;
  };
  const size_t NB = (size_t)NN * DD;
  u16*   xph  = (u16*)take(NB * 2);
  u16*   xpl  = (u16*)take(NB * 2);
  u16*   t1h  = (u16*)take(NB * 2);
  u16*   t1l  = (u16*)take(NB * 2);
  float* hyp0 = (float*)take(NB * 4);
  float* ABuf = (float*)take((size_t)NN * ABW * 4);
  float* msg  = (float*)take(NB * 4);
  float* agg  = (float*)take(NB * 4);
  float* h1   = (float*)take(NB * 4);
  float* ew   = (float*)take((size_t)NE * 4);
  u16*   w1h  = (u16*)take((size_t)ABW * DD * 2);
  u16*   w1l  = (u16*)take((size_t)ABW * DD * 2);
  u16*   gh   = (u16*)take((size_t)DD * DD * 2);
  u16*   gl   = (u16*)take((size_t)DD * DD * 2);
  float* cst  = (float*)take(1024 * 4);
  if (off > ws_size) return;  // scratch too small; bail deterministically

  const int PW = (NN + 7) / 8;          // 12500 row-wave blocks
  const int EB = (NE + 7) / 8;          // 62500 edge-wave blocks
  const int GB = (NN + 127) / 128;      // 782 GEMM row-blocks

  init_nodes<<<PW, 256, 0, stream>>>(x_proj, xph, xpl, hyp0, NN);

  for (int v = 0; v < VV; ++v) {
    const float* W1v = rw_W1 + (size_t)v * HIDN * 291;
    const float* b1v = rw_b1 + (size_t)v * HIDN;
    const float* W2v = rw_W2 + (size_t)v * HIDN;
    const float* b2v = rw_b2 + v;
    const float* g1v = g1W + (size_t)v * DD * DD;
    const float* g2v = g2W + (size_t)v * DD * DD;
    const int*   srcI = eidx + (size_t)v * 2 * NE;
    const int*   dstI = srcI + NE;
    const float* escv = edge_scores + (size_t)v * NE;

    // ---- rewiring: node projections + per-edge MLP ----
    conv_w1ab<<<(ABW * DD + 255) / 256, 256, 0, stream>>>(W1v, w1h, w1l);
    rewire_consts<<<1, 320, 0, stream>>>(W1v, b1v, W2v, b2v, ctab, ctid, cst);
    gemm_bf16x3<<<dim3(GB, ABW / 16), 256, 0, stream>>>(xph, xpl, w1h, w1l,
                                                        ABuf, NN, ABW, cst, causal);
    edge_rewire<<<EB, 256, 0, stream>>>(ABuf, cst, srcI, dstI, escv, ew, NE);

    // ---- GCN layer 1 (identity = hyp0) ----
    conv_w128<<<(DD * DD + 255) / 256, 256, 0, stream>>>(g1v, gh, gl);
    logmap_bf16<<<PW, 256, 0, stream>>>(hyp0, t1h, t1l, NN);
    gemm_bf16x3<<<dim3(GB, DD / 16), 256, 0, stream>>>(t1h, t1l, gh, gl,
                                                       msg, NN, DD, nullptr, nullptr);
    msg_transform<<<PW, 256, 0, stream>>>(msg, agg, NN);
    edge_scatter<<<EB, 256, 0, stream>>>(msg, srcI, dstI, ew, agg, NE);
    update_nodes<<<PW, 256, 0, stream>>>(agg, hyp0, h1, nullptr, 0, 0, NN);

    // ---- GCN layer 2 (identity = h1); writes hyp + tan outputs ----
    conv_w128<<<(DD * DD + 255) / 256, 256, 0, stream>>>(g2v, gh, gl);
    logmap_bf16<<<PW, 256, 0, stream>>>(h1, t1h, t1l, NN);
    gemm_bf16x3<<<dim3(GB, DD / 16), 256, 0, stream>>>(t1h, t1l, gh, gl,
                                                       msg, NN, DD, nullptr, nullptr);
    msg_transform<<<PW, 256, 0, stream>>>(msg, agg, NN);
    edge_scatter<<<EB, 256, 0, stream>>>(msg, srcI, dstI, ew, agg, NE);
    float* hypOut = out + (size_t)NN * VV * DD + (size_t)v * NN * DD;
    update_nodes<<<PW, 256, 0, stream>>>(agg, h1, hypOut, out, VV * DD, v * DD, NN);
  }
}
